// CaptchaRecognizer_31696858645147
// MI455X (gfx1250) — matrix operand order repaired and checked by reading, not compiled, not run on hardware
//
#include <hip/hip_runtime.h>
#include <hip/hip_bf16.h>

typedef __attribute__((ext_vector_type(16))) _Float16 v16h;
typedef __attribute__((ext_vector_type(8)))  _Float16 v8h;
typedef __attribute__((ext_vector_type(8)))  float    v8f;

// ---------------- model constants ----------------
constexpr int   BATCH = 32;
constexpr int   TSTEP = 32;
constexpr int   IMGH = 80, IMGW = 200;
constexpr int   FH = 14, FW = 44;
constexpr int   NH0 = 192, NH1 = 128, NOUT = 37, NOUTP = 48;
constexpr int   MROW = BATCH * FW;          // 1408
constexpr int   FEAT = 32 * FH;             // 448
constexpr float DT_TAUMEM = 0.1f;           // DT * 100
constexpr float DT_TAUSYN = 0.2f;           // DT * 200
constexpr float DT_TAUADP = 0.001f / 0.7f;
constexpr float ADP_BETA  = 1.8f / 0.7f;

// ---------------- WMMA fragment helpers ----------------
// A 16x32 f16 row-major, lda halfs.  lane holds row = (lane&15);
// halfs 0..7  -> K = k0 + 8*(lane>>4) + 0..7
// halfs 8..15 -> K = k0 + 8*(lane>>4) + 16..23
__device__ __forceinline__ v16h load_a_frag(const _Float16* __restrict__ A, int lda,
                                            int row, int k0, int lane) {
  const _Float16* p = A + (size_t)row * lda + k0 + 8 * (lane >> 4);
  v8h c0 = *(const v8h*)(p);
  v8h c1 = *(const v8h*)(p + 16);
  return __builtin_shufflevector(c0, c1, 0, 1, 2, 3, 4, 5, 6, 7,
                                 8, 9, 10, 11, 12, 13, 14, 15);
}

// B 32x16 f16 from row-major weight W[N, K], ldw halfs. lane holds col = (lane&15);
// halfs 0..7  -> K = k0 + 8*(lane>>4) + 0..7
// halfs 8..15 -> K = k0 + 8*(lane>>4) + 16..23
__device__ __forceinline__ v16h load_b_frag(const _Float16* __restrict__ W, int ldw,
                                            int col, int k0, int lane) {
  const _Float16* p = W + (size_t)col * ldw + k0 + 8 * (lane >> 4);
  v8h c0 = *(const v8h*)(p);
  v8h c1 = *(const v8h*)(p + 16);
  return __builtin_shufflevector(c0, c1, 0, 1, 2, 3, 4, 5, 6, 7,
                                 8, 9, 10, 11, 12, 13, 14, 15);
}

__device__ __forceinline__ v8f wmma_f16(v16h a, v16h b, v8f c) {
  return __builtin_amdgcn_wmma_f32_16x16x32_f16(false, a, false, b, (short)0, c, false, false);
}

// compile-time NHWC im2col offset for logical K index k (order: kh, kw, cin — cin fastest)
template <int CIN, int KS, int WW>
__host__ __device__ constexpr int imoff_nhwc(int k) {
  int khkw = k / CIN;
  int cin = k - khkw * CIN;
  int kh = khkw / KS;
  int kw = khkw - kh * KS;
  return (kh * WW + kw) * CIN + cin;
}

// ---------------- encoder (ConstantCurrentLIF) ----------------
__global__ void encode_step_k(const float* __restrict__ img, float* __restrict__ v,
                              _Float16* __restrict__ spk, int n) {
  int i = blockIdx.x * blockDim.x + threadIdx.x;
  if (i >= n) return;
  float vv = v[i];
  vv += DT_TAUMEM * (img[i] - vv);
  float z = (vv - 1.0f) > 0.0f ? 1.0f : 0.0f;
  v[i] = vv * (1.0f - z);
  spk[i] = (_Float16)z;
}

// ---------------- conv1 (1->6, k3) + BN + pool2x2, direct; NHWC output ----------------
__global__ void conv1_bn_pool_k(const _Float16* __restrict__ spk, const float* __restrict__ w,
                                const float* __restrict__ sc, const float* __restrict__ sh,
                                _Float16* __restrict__ out) {
  constexpr int OH = 39, OW = 99;
  int idx = blockIdx.x * blockDim.x + threadIdx.x;
  int n = BATCH * OH * OW * 6;
  if (idx >= n) return;
  int co = idx % 6;  int t = idx / 6;
  int ow = t % OW;   t /= OW;
  int oh = t % OH;   int b = t / OH;
  const _Float16* sp = spk + (size_t)b * IMGH * IMGW;
  const float* wc = w + co * 9;
  float mx = -3.0e38f;
#pragma unroll
  for (int dy = 0; dy < 2; ++dy)
#pragma unroll
    for (int dx = 0; dx < 2; ++dx) {
      int ih = 2 * oh + dy, iw = 2 * ow + dx;
      float s = 0.f;
#pragma unroll
      for (int kh = 0; kh < 3; ++kh)
#pragma unroll
        for (int kw = 0; kw < 3; ++kw)
          s += (float)sp[(ih + kh) * IMGW + iw + kw] * wc[kh * 3 + kw];
      float val = s * sc[co] + sh[co];
      mx = fmaxf(mx, val);
    }
  out[idx] = (_Float16)mx;   // NHWC: ((b*OH+oh)*OW+ow)*6+co
}

// ---------------- implicit-GEMM conv + BN (WMMA), NHWC in / NHWC out ----------------
// M = BATCH*OH*OW output pixels, N = COUT (padded to COUTP), K = KS*KS*CIN (padded KPAD).
// Input buffer is over-allocated+zeroed so reads up to imoff_nhwc(KPAD-1) stay in bounds;
// padded weight columns are zero so those terms contribute nothing.
template <int CIN, int HH, int WW, int COUT, int COUTP, int OH, int OW, int KS, int KPAD>
__global__ void conv_gemm_bn_k(const _Float16* __restrict__ in, const _Float16* __restrict__ Wt,
                               const float* __restrict__ sc, const float* __restrict__ sh,
                               _Float16* __restrict__ out) {
  constexpr int M = BATCH * OH * OW;
  constexpr int KT = KPAD / 32;
  constexpr int NT = COUTP / 16;
  const int lane = threadIdx.x & 31;
  const int mt = blockIdx.x * 4 + (threadIdx.x >> 5);
  if (mt * 16 >= M) return;
  const int hi = lane >> 4;
  const int l15 = lane & 15;

  const int m = mt * 16 + l15;
  const int bimg = m / (OH * OW);
  const int p = m - bimg * (OH * OW);
  const int oh = p / OW;
  const int ow = p - oh * OW;
  const _Float16* base2 = in + ((size_t)bimg * HH * WW + oh * WW + ow) * CIN;

  v8f acc[NT];
#pragma unroll
  for (int n = 0; n < NT; ++n) acc[n] = v8f{};

#pragma unroll
  for (int kt = 0; kt < KT; ++kt) {
    v16h a;
    if (hi == 0) {
#pragma unroll
      for (int e = 0; e < 16; ++e) {
        const int k = kt * 32 + (e & 7) + ((e >= 8) ? 16 : 0);
        a[e] = base2[imoff_nhwc<CIN, KS, WW>(k)];
      }
    } else {
#pragma unroll
      for (int e = 0; e < 16; ++e) {
        const int k = kt * 32 + 8 + (e & 7) + ((e >= 8) ? 16 : 0);
        a[e] = base2[imoff_nhwc<CIN, KS, WW>(k)];
      }
    }
#pragma unroll
    for (int n = 0; n < NT; ++n) {
      v16h b = load_b_frag(Wt, KPAD, n * 16 + l15, kt * 32, lane);
      acc[n] = wmma_f16(a, b, acc[n]);
    }
  }

  // NHWC store: row index of GEMM == pixel index, so address = m*COUT + col
#pragma unroll
  for (int n = 0; n < NT; ++n) {
    const int col = n * 16 + l15;
    if (col < COUT) {
      float s = sc[col], h = sh[col];
      _Float16* ob = out + ((size_t)mt * 16 + 8 * hi) * COUT + col;
#pragma unroll
      for (int r = 0; r < 8; ++r)
        ob[r * COUT] = (_Float16)(acc[n][r] * s + h);
    }
  }
}

// ---------------- linear GEMM: out = X@WX.T (+ Z@WZ.T), WMMA ----------------
// Each wave computes one 16-row M tile x NT 16-col N tiles (A-fragment reuse).
template <int KX, int KZ, int NT>
__global__ void lin_gemm_k(const _Float16* __restrict__ X, const _Float16* __restrict__ WX,
                           const _Float16* __restrict__ Z, const _Float16* __restrict__ WZ,
                           float* __restrict__ out, int M, int Npad) {
  const int lane = threadIdx.x & 31;
  const int mt = blockIdx.x * 4 + (threadIdx.x >> 5);
  if (mt * 16 >= M) return;
  const int l15 = lane & 15;
  const int row = mt * 16 + l15;
  const int ntb = blockIdx.y * NT;

  v8f acc[NT];
#pragma unroll
  for (int n = 0; n < NT; ++n) acc[n] = v8f{};

#pragma unroll
  for (int kt = 0; kt < KX / 32; ++kt) {
    v16h a = load_a_frag(X, KX, row, kt * 32, lane);
#pragma unroll
    for (int n = 0; n < NT; ++n)
      acc[n] = wmma_f16(a, load_b_frag(WX, KX, (ntb + n) * 16 + l15, kt * 32, lane), acc[n]);
  }
#pragma unroll
  for (int kt = 0; kt < KZ / 32; ++kt) {
    v16h a = load_a_frag(Z, KZ, row, kt * 32, lane);
#pragma unroll
    for (int n = 0; n < NT; ++n)
      acc[n] = wmma_f16(a, load_b_frag(WZ, KZ, (ntb + n) * 16 + l15, kt * 32, lane), acc[n]);
  }

  const int hi = lane >> 4;
#pragma unroll
  for (int n = 0; n < NT; ++n) {
    const int col = (ntb + n) * 16 + l15;
#pragma unroll
    for (int r = 0; r < 8; ++r)
      out[(size_t)(mt * 16 + r + 8 * hi) * Npad + col] = acc[n][r];
  }
}

// ---------------- LSNN pointwise state update ----------------
__global__ void pw_lsnn_k(float* __restrict__ v, float* __restrict__ i_, float* __restrict__ b,
                          _Float16* __restrict__ z, const float* __restrict__ gm, int n) {
  int idx = blockIdx.x * blockDim.x + threadIdx.x;
  if (idx >= n) return;
  float vv = v[idx], ii = i_[idx], bb = b[idx];
  float v_dec = vv + DT_TAUMEM * (ii - vv);
  float i_dec = ii - DT_TAUSYN * ii;
  float b_dec = bb + DT_TAUADP * (1.0f - bb);
  float zn = (v_dec - b_dec) > 0.0f ? 1.0f : 0.0f;
  v[idx] = (1.0f - zn) * v_dec;
  i_[idx] = i_dec + gm[idx];
  b[idx] = b_dec + zn * ADP_BETA;
  z[idx] = (_Float16)zn;
}

// ---------------- LI pointwise + running max over T ----------------
__global__ void pw_li_k(float* __restrict__ v, float* __restrict__ i_, float* __restrict__ vmax,
                        const float* __restrict__ gm, int n) {
  int idx = blockIdx.x * blockDim.x + threadIdx.x;
  if (idx >= n) return;
  int m = idx / NOUT, c = idx - m * NOUT;
  float vv = v[idx], ii = i_[idx];
  float vn = vv + DT_TAUMEM * (ii - vv);       // uses OLD i (norse li_step)
  i_[idx] = ii - DT_TAUSYN * ii + gm[(size_t)m * NOUTP + c];
  v[idx] = vn;
  vmax[idx] = fmaxf(vmax[idx], vn);
}

// ---------------- pool(2x2) of conv5 (NHWC) + reshape to [B, FW, 448] ----------------
__global__ void pool5_fm_k(const _Float16* __restrict__ a5, _Float16* __restrict__ fm) {
  constexpr int OH = 29, OW = 89;
  int idx = blockIdx.x * blockDim.x + threadIdx.x;
  int n = BATCH * FW * FEAT;
  if (idx >= n) return;
  int f = idx % FEAT; int t = idx / FEAT;
  int w = t % FW;     int b = t / FW;
  int c = f / FH, h = f - c * FH;
  const _Float16* p = a5 + (size_t)b * OH * OW * 32 + c;
  float mx = -3.0e38f;
#pragma unroll
  for (int dy = 0; dy < 2; ++dy)
#pragma unroll
    for (int dx = 0; dx < 2; ++dx)
      mx = fmaxf(mx, (float)p[((2 * h + dy) * OW + 2 * w + dx) * 32]);
  fm[idx] = (_Float16)mx;
}

// ---------------- prep / init ----------------
__global__ void prep_bn_k(const float* g, const float* b, const float* m, const float* v,
                          float* sc, float* sh, int C) {
  int i = blockIdx.x * blockDim.x + threadIdx.x;
  if (i >= C) return;
  float s = g[i] * rsqrtf(v[i] + 1e-5f);
  sc[i] = s;
  sh[i] = b[i] - m[i] * s;
}

// linear weight: f32 [O,I] -> f16 [Opad,Ipad] zero-padded
__global__ void pack_w_k(const float* __restrict__ src, _Float16* __restrict__ dst,
                         int O, int I, int Opad, int Ipad) {
  int idx = blockIdx.x * blockDim.x + threadIdx.x;
  int n = Opad * Ipad;
  if (idx >= n) return;
  int i = idx % Ipad, o = idx / Ipad;
  dst[idx] = (o < O && i < I) ? (_Float16)src[(size_t)o * I + i] : (_Float16)0.f;
}

// conv weight: f32 [COUT,CIN,KS,KS] -> f16 [COUTP, KPAD] with K order (kh,kw,cin)
__global__ void pack_convw_k(const float* __restrict__ src, _Float16* __restrict__ dst,
                             int COUT, int CIN, int KS, int COUTP, int KPAD) {
  int idx = blockIdx.x * blockDim.x + threadIdx.x;
  int n = COUTP * KPAD;
  if (idx >= n) return;
  int k = idx % KPAD, o = idx / KPAD;
  int kreal = KS * KS * CIN;
  _Float16 val = (_Float16)0.f;
  if (o < COUT && k < kreal) {
    int khkw = k / CIN;
    int cin = k - khkw * CIN;
    val = (_Float16)src[(size_t)o * CIN * KS * KS + cin * KS * KS + khkw];
  }
  dst[idx] = val;
}

__global__ void fill_f32_k(float* p, float v, int n) {
  int i = blockIdx.x * blockDim.x + threadIdx.x;
  if (i < n) p[i] = v;
}
__global__ void fill_f16_k(_Float16* p, float v, int n) {
  int i = blockIdx.x * blockDim.x + threadIdx.x;
  if (i < n) p[i] = (_Float16)v;
}

__global__ void finalize_k(const float* __restrict__ vmax, void* __restrict__ dout) {
  int idx = blockIdx.x * blockDim.x + threadIdx.x;
  constexpr int NV = MROW * NOUT;          // 52096 voltages
  if (idx < NV)              ((float*)dout)[idx] = vmax[idx];
  else if (idx < NV + BATCH) ((int*)dout)[idx] = FW;   // voltages_length = 44
}

// ---------------- host launcher ----------------
extern "C" void kernel_launch(void* const* d_in, const int* in_sizes, int n_in,
                              void* d_out, int out_size, void* d_ws, size_t ws_size,
                              hipStream_t stream) {
  (void)in_sizes; (void)n_in; (void)out_size; (void)ws_size;
  const float* images = (const float*)d_in[0];
  const float* w1 = (const float*)d_in[1];
  const float* g1 = (const float*)d_in[2];  const float* b1 = (const float*)d_in[3];
  const float* m1 = (const float*)d_in[4];  const float* vv1 = (const float*)d_in[5];
  const float* w2 = (const float*)d_in[6];
  const float* g2 = (const float*)d_in[7];  const float* b2 = (const float*)d_in[8];
  const float* m2 = (const float*)d_in[9];  const float* vv2 = (const float*)d_in[10];
  const float* w3 = (const float*)d_in[11];
  const float* g3 = (const float*)d_in[12]; const float* b3 = (const float*)d_in[13];
  const float* m3 = (const float*)d_in[14]; const float* vv3 = (const float*)d_in[15];
  const float* w4 = (const float*)d_in[16];
  const float* g4 = (const float*)d_in[17]; const float* b4 = (const float*)d_in[18];
  const float* m4 = (const float*)d_in[19]; const float* vv4 = (const float*)d_in[20];
  const float* w5 = (const float*)d_in[21];
  const float* g5 = (const float*)d_in[22]; const float* b5 = (const float*)d_in[23];
  const float* m5 = (const float*)d_in[24]; const float* vv5 = (const float*)d_in[25];
  const float* w_in0  = (const float*)d_in[26];
  const float* w_rec0 = (const float*)d_in[27];
  const float* w_in1  = (const float*)d_in[28];
  const float* w_rec1 = (const float*)d_in[29];
  const float* w_out  = (const float*)d_in[30];

  char* ws = (char*)d_ws;
  size_t off = 0;
  auto alloc = [&](size_t bytes) -> char* {
    off = (off + 255) & ~(size_t)255;
    char* p = ws + off;
    off += bytes;
    return p;
  };

  constexpr int NPIX = BATCH * IMGH * IMGW;          // 512000
  constexpr int NA1 = BATCH * 39 * 99 * 6;           // 741312  (NHWC)
  constexpr int NA2 = BATCH * 37 * 97 * 12;          // 1378176
  constexpr int NA3 = BATCH * 33 * 93 * 16;          // 1571328
  constexpr int NA4 = BATCH * 31 * 91 * 24;          // 2166528
  constexpr int NA5 = BATCH * 29 * 89 * 32;          // 2642944
  // gather over-read padding (monotone offset => max at k = KPAD-1)
  constexpr int PAD1 = imoff_nhwc<6, 3, 99>(64 - 1) + 16;    // conv2 reads act1
  constexpr int PAD2 = imoff_nhwc<12, 5, 97>(320 - 1) + 16;  // conv3 reads act2
  constexpr int PAD3 = imoff_nhwc<16, 3, 93>(160 - 1) + 16;  // conv4 reads act3
  constexpr int PAD4 = imoff_nhwc<24, 3, 91>(224 - 1) + 16;  // conv5 reads act4
  constexpr int NFM = MROW * FEAT;                   // 630784
  constexpr int NS0 = MROW * NH0;                    // 270336
  constexpr int NS1 = MROW * NH1;                    // 180224
  constexpr int NSO = MROW * NOUT;                   // 52096

  float*    v_enc = (float*)   alloc((size_t)NPIX * 4);
  _Float16* spk   = (_Float16*)alloc((size_t)NPIX * 2);
  _Float16* act1  = (_Float16*)alloc((size_t)(NA1 + PAD1) * 2);
  _Float16* act2  = (_Float16*)alloc((size_t)(NA2 + PAD2) * 2);
  _Float16* act3  = (_Float16*)alloc((size_t)(NA3 + PAD3) * 2);
  _Float16* act4  = (_Float16*)alloc((size_t)(NA4 + PAD4) * 2);
  _Float16* act5  = (_Float16*)alloc((size_t)NA5 * 2);
  _Float16* fm    = (_Float16*)alloc((size_t)NFM * 2);
  _Float16* z0    = (_Float16*)alloc((size_t)NS0 * 2);
  float* v0 = (float*)alloc((size_t)NS0 * 4);
  float* i0 = (float*)alloc((size_t)NS0 * 4);
  float* bb0 = (float*)alloc((size_t)NS0 * 4);
  _Float16* z1 = (_Float16*)alloc((size_t)NS1 * 2);
  float* v1 = (float*)alloc((size_t)NS1 * 4);
  float* i1 = (float*)alloc((size_t)NS1 * 4);
  float* bb1 = (float*)alloc((size_t)NS1 * 4);
  float* vli  = (float*)alloc((size_t)NSO * 4);
  float* ili  = (float*)alloc((size_t)NSO * 4);
  float* vmax = (float*)alloc((size_t)NSO * 4);
  float* tmp  = (float*)alloc((size_t)MROW * NH0 * 4);   // reused GEMM scratch
  _Float16* wc2 = (_Float16*)alloc(16 * 64 * 2);
  _Float16* wc3 = (_Float16*)alloc(16 * 320 * 2);
  _Float16* wc4 = (_Float16*)alloc(32 * 160 * 2);
  _Float16* wc5 = (_Float16*)alloc(32 * 224 * 2);
  float* sc1 = (float*)alloc(6 * 4);  float* sh1 = (float*)alloc(6 * 4);
  float* sc2 = (float*)alloc(12 * 4); float* sh2 = (float*)alloc(12 * 4);
  float* sc3 = (float*)alloc(16 * 4); float* sh3 = (float*)alloc(16 * 4);
  float* sc4 = (float*)alloc(24 * 4); float* sh4 = (float*)alloc(24 * 4);
  float* sc5 = (float*)alloc(32 * 4); float* sh5 = (float*)alloc(32 * 4);
  _Float16* win0  = (_Float16*)alloc((size_t)NH0 * FEAT * 2);
  _Float16* wrec0 = (_Float16*)alloc((size_t)NH0 * NH0 * 2);
  _Float16* win1  = (_Float16*)alloc((size_t)NH1 * NH0 * 2);
  _Float16* wrec1 = (_Float16*)alloc((size_t)NH1 * NH1 * 2);
  _Float16* wout  = (_Float16*)alloc((size_t)NOUTP * NH1 * 2);

  auto cdiv = [](int a, int b) { return (a + b - 1) / b; };

  // ---- weight prep: BN fold + f16 packing ----
  prep_bn_k<<<1, 64, 0, stream>>>(g1, b1, m1, vv1, sc1, sh1, 6);
  prep_bn_k<<<1, 64, 0, stream>>>(g2, b2, m2, vv2, sc2, sh2, 12);
  prep_bn_k<<<1, 64, 0, stream>>>(g3, b3, m3, vv3, sc3, sh3, 16);
  prep_bn_k<<<1, 64, 0, stream>>>(g4, b4, m4, vv4, sc4, sh4, 24);
  prep_bn_k<<<1, 64, 0, stream>>>(g5, b5, m5, vv5, sc5, sh5, 32);
  pack_convw_k<<<cdiv(16 * 64, 256), 256, 0, stream>>>(w2, wc2, 12, 6, 3, 16, 64);
  pack_convw_k<<<cdiv(16 * 320, 256), 256, 0, stream>>>(w3, wc3, 16, 12, 5, 16, 320);
  pack_convw_k<<<cdiv(32 * 160, 256), 256, 0, stream>>>(w4, wc4, 24, 16, 3, 32, 160);
  pack_convw_k<<<cdiv(32 * 224, 256), 256, 0, stream>>>(w5, wc5, 32, 24, 3, 32, 224);
  pack_w_k<<<cdiv(NH0 * FEAT, 256), 256, 0, stream>>>(w_in0, win0, NH0, FEAT, NH0, FEAT);
  pack_w_k<<<cdiv(NH0 * NH0, 256), 256, 0, stream>>>(w_rec0, wrec0, NH0, NH0, NH0, NH0);
  pack_w_k<<<cdiv(NH1 * NH0, 256), 256, 0, stream>>>(w_in1, win1, NH1, NH0, NH1, NH0);
  pack_w_k<<<cdiv(NH1 * NH1, 256), 256, 0, stream>>>(w_rec1, wrec1, NH1, NH1, NH1, NH1);
  pack_w_k<<<cdiv(NOUTP * NH1, 256), 256, 0, stream>>>(w_out, wout, NOUT, NH1, NOUTP, NH1);

  // ---- state init (zero act buffers once so gather padding reads finite zeros) ----
  fill_f32_k<<<cdiv(NPIX, 256), 256, 0, stream>>>(v_enc, 0.f, NPIX);
  fill_f16_k<<<cdiv(NA1 + PAD1, 256), 256, 0, stream>>>(act1, 0.f, NA1 + PAD1);
  fill_f16_k<<<cdiv(NA2 + PAD2, 256), 256, 0, stream>>>(act2, 0.f, NA2 + PAD2);
  fill_f16_k<<<cdiv(NA3 + PAD3, 256), 256, 0, stream>>>(act3, 0.f, NA3 + PAD3);
  fill_f16_k<<<cdiv(NA4 + PAD4, 256), 256, 0, stream>>>(act4, 0.f, NA4 + PAD4);
  fill_f16_k<<<cdiv(NS0, 256), 256, 0, stream>>>(z0, 0.f, NS0);
  fill_f32_k<<<cdiv(NS0, 256), 256, 0, stream>>>(v0, 0.f, NS0);
  fill_f32_k<<<cdiv(NS0, 256), 256, 0, stream>>>(i0, 0.f, NS0);
  fill_f32_k<<<cdiv(NS0, 256), 256, 0, stream>>>(bb0, 1.0f, NS0);
  fill_f16_k<<<cdiv(NS1, 256), 256, 0, stream>>>(z1, 0.f, NS1);
  fill_f32_k<<<cdiv(NS1, 256), 256, 0, stream>>>(v1, 0.f, NS1);
  fill_f32_k<<<cdiv(NS1, 256), 256, 0, stream>>>(i1, 0.f, NS1);
  fill_f32_k<<<cdiv(NS1, 256), 256, 0, stream>>>(bb1, 1.0f, NS1);
  fill_f32_k<<<cdiv(NSO, 256), 256, 0, stream>>>(vli, 0.f, NSO);
  fill_f32_k<<<cdiv(NSO, 256), 256, 0, stream>>>(ili, 0.f, NSO);
  fill_f32_k<<<cdiv(NSO, 256), 256, 0, stream>>>(vmax, -3.0e38f, NSO);

  // ---- time loop ----
  for (int t = 0; t < TSTEP; ++t) {
    encode_step_k<<<cdiv(NPIX, 256), 256, 0, stream>>>(images, v_enc, spk, NPIX);
    conv1_bn_pool_k<<<cdiv(NA1, 256), 256, 0, stream>>>(spk, w1, sc1, sh1, act1);
    conv_gemm_bn_k<6, 39, 99, 12, 16, 37, 97, 3, 64>
        <<<cdiv(BATCH * 37 * 97 / 16, 4), 128, 0, stream>>>(act1, wc2, sc2, sh2, act2);
    conv_gemm_bn_k<12, 37, 97, 16, 16, 33, 93, 5, 320>
        <<<cdiv(BATCH * 33 * 93 / 16, 4), 128, 0, stream>>>(act2, wc3, sc3, sh3, act3);
    conv_gemm_bn_k<16, 33, 93, 24, 32, 31, 91, 3, 160>
        <<<cdiv(BATCH * 31 * 91 / 16, 4), 128, 0, stream>>>(act3, wc4, sc4, sh4, act4);
    conv_gemm_bn_k<24, 31, 91, 32, 32, 29, 89, 3, 224>
        <<<cdiv(BATCH * 29 * 89 / 16, 4), 128, 0, stream>>>(act4, wc5, sc5, sh5, act5);
    pool5_fm_k<<<cdiv(NFM, 256), 256, 0, stream>>>(act5, fm);

    // LSNN layer 0: tmp = fm@Win0.T + z0@Wrec0.T  (uses OLD z0), then state update
    lin_gemm_k<FEAT, NH0, 2><<<dim3(MROW / 16 / 4, NH0 / 16 / 2), 128, 0, stream>>>(
        fm, win0, z0, wrec0, tmp, MROW, NH0);
    pw_lsnn_k<<<cdiv(NS0, 256), 256, 0, stream>>>(v0, i0, bb0, z0, tmp, NS0);
    // LSNN layer 1
    lin_gemm_k<NH0, NH1, 2><<<dim3(MROW / 16 / 4, NH1 / 16 / 2), 128, 0, stream>>>(
        z0, win1, z1, wrec1, tmp, MROW, NH1);
    pw_lsnn_k<<<cdiv(NS1, 256), 256, 0, stream>>>(v1, i1, bb1, z1, tmp, NS1);
    // LI readout (NT=3 covers all 48 padded cols in one pass)
    lin_gemm_k<NH1, 0, 3><<<dim3(MROW / 16 / 4, 1), 128, 0, stream>>>(
        z1, wout, z1, wout, tmp, MROW, NOUTP);
    pw_li_k<<<cdiv(NSO, 256), 256, 0, stream>>>(vli, ili, vmax, tmp, NSO);
  }

  finalize_k<<<cdiv(NSO + BATCH, 256), 256, 0, stream>>>(vmax, d_out);
}
